// Transformer_33878702031436
// MI455X (gfx1250) — matrix operand order repaired and checked by reading, not compiled, not run on hardware
//
#include <hip/hip_runtime.h>
#include <hip/hip_bf16.h>
#include <math.h>

typedef __attribute__((ext_vector_type(16))) _Float16 v16h;
typedef __attribute__((ext_vector_type(8)))  _Float16 v8h;
typedef __attribute__((ext_vector_type(8)))  float    v8f;

// ---------------- problem constants ----------------
#define BATCH 8192
#define SEQ   16
#define DIM   128
#define HEADS 8
#define HID   1024
#define GCN   40
#define NUMF  103
#define EPS   1e-5f

// ---------------- workspace layout (bytes) ----------------
#define OFF_PE    0
#define OFF_MU    8192
#define OFF_RSTD  8704
#define OFF_PSUM  16384
#define OFF_PSQ   81920
#define OFF_W     147456
// half-element offsets inside fragment region
#define HQ  0
#define HK  32768
#define HV  65536
#define HO  98304
#define H1  131072
#define H2  393216

static __device__ __forceinline__ v8f wmma16(v16h a, v16h b, v8f c) {
    return __builtin_amdgcn_wmma_f32_16x16x32_f16(false, a, false, b,
                                                  (short)0, c, false, false);
}

// A fragment (16xK f16) from LDS, row stride ld halfs, k-tile base k0.
static __device__ __forceinline__ v16h load_a_frag(const _Float16* base, int lane,
                                                   int k0, int ld) {
    int m    = lane & 15;
    int koff = (lane >> 4) * 8;
    const _Float16* p = base + m * ld + k0 + koff;
    v8h lo = *(const v8h*)(p);
    v8h hi = *(const v8h*)(p + 16);
    v16h a;
#pragma unroll
    for (int i = 0; i < 8; ++i) { a[i] = lo[i]; a[i + 8] = hi[i]; }
    return a;
}

// B fragment from pre-swizzled global weights: 16 contiguous halfs per lane
static __device__ __forceinline__ v16h load_b_frag(const _Float16* wfrag, int lane,
                                                   int tilen, int tilek, int ktiles) {
    const _Float16* p = wfrag + (((size_t)(tilen * ktiles + tilek)) * 32 + lane) * 16;
    return *(const v16h*)p;
}

// B fragment from LDS staging buffer (frag-major, 512 halfs per fragment)
static __device__ __forceinline__ v16h load_b_frag_lds(const _Float16* base, int lane,
                                                       int frag) {
    return *(const v16h*)(base + frag * 512 + lane * 16);
}

// async global->LDS copy: 16B per lane (512B per wave-instruction). ASYNCcnt.
static __device__ __forceinline__ void async_ld_b128(unsigned lds_off, const void* gaddr) {
    asm volatile("global_load_async_to_lds_b128 %0, %1, off"
                 :: "v"(lds_off), "v"(gaddr) : "memory");
}
static __device__ __forceinline__ void wait_async0() {
    asm volatile("s_wait_asynccnt 0" ::: "memory");
}

static __device__ __forceinline__ float wave_sum(float v) {
#pragma unroll
    for (int o = 16; o > 0; o >>= 1) v += __shfl_xor(v, o, 32);
    return v;
}

// D=128 -> D=128 GEMM: 8 n-tiles x 4 k-steps, A preloaded in regs.
static __device__ __forceinline__ void gemm128(const v16h* areg, const _Float16* wfrag,
                                               const float* bias, float* outF,
                                               _Float16* outH, int lane) {
    int n  = lane & 15;
    int mb = (lane >> 4) * 8;
#pragma unroll
    for (int nt = 0; nt < 8; ++nt) {
        v8f c = {};
#pragma unroll
        for (int kt = 0; kt < 4; ++kt)
            c = wmma16(areg[kt], load_b_frag(wfrag, lane, nt, kt, 4), c);
        float bv = bias[nt * 16 + n];
        if (outF) {
#pragma unroll
            for (int r = 0; r < 8; ++r)
                outF[(mb + r) * DIM + nt * 16 + n] = c[r] + bv;
        }
        if (outH) {
#pragma unroll
            for (int r = 0; r < 8; ++r)
                outH[(mb + r) * DIM + nt * 16 + n] = (_Float16)(c[r] + bv);
        }
    }
}

// LayerNorm over D of (acc + cur); writes cur (f32) and actH (f16)
static __device__ __forceinline__ void ln_res(const float* accL, float* curL,
                                              const float* g, const float* b,
                                              _Float16* actH, int lane) {
    int c0 = lane * 4;
    for (int r = 0; r < 16; ++r) {
        float v[4];
        float s = 0.f;
#pragma unroll
        for (int j = 0; j < 4; ++j) {
            v[j] = accL[r * DIM + c0 + j] + curL[r * DIM + c0 + j];
            s += v[j];
        }
        s = wave_sum(s);
        float mean = s * (1.0f / DIM);
        float q = 0.f;
#pragma unroll
        for (int j = 0; j < 4; ++j) { float d = v[j] - mean; q += d * d; }
        q = wave_sum(q);
        float rs = rsqrtf(q * (1.0f / DIM) + EPS);
#pragma unroll
        for (int j = 0; j < 4; ++j) {
            float o = (v[j] - mean) * rs * g[c0 + j] + b[c0 + j];
            curL[r * DIM + c0 + j] = o;
            actH[r * DIM + c0 + j] = (_Float16)o;
        }
    }
}

// ---------------- prep kernels ----------------
__global__ void pe_kernel(float* pe) {
    int i = blockIdx.x * blockDim.x + threadIdx.x;
    if (i >= SEQ * DIM) return;
    int s = i >> 7, d = i & 127;
    float ang = (float)s / powf(10000.0f, (float)(d & ~1) / (float)DIM);
    pe[i] = (d & 1) ? cosf(ang) : sinf(ang);
}

__global__ void cvt_frag(const float* __restrict__ src, _Float16* __restrict__ dst,
                         int K, int N, int total) {
    int idx = blockIdx.x * blockDim.x + threadIdx.x;
    if (idx >= total) return;
    int perLayer = K * N;
    int layer = idx / perLayer;
    int o = idx - layer * perLayer;
    int e = o & 15;
    int L = (o >> 4) & 31;
    int t = o >> 9;
    int kt = K / 32;
    int tilek = t % kt;
    int tilen = t / kt;
    int k = tilek * 32 + (L >> 4) * 8 + (e < 8 ? e : e + 8);
    int n = tilen * 16 + (L & 15);
    dst[idx] = (_Float16)src[layer * perLayer + k * N + n];
}

__global__ void bn_stats1(const float* __restrict__ num, const float* __restrict__ Wn,
                          const float* __restrict__ bnum,
                          float* __restrict__ psum, float* __restrict__ psq) {
    int f  = threadIdx.x & 127;
    int g  = threadIdx.x >> 7;
    int b0 = blockIdx.x * 128 + g * 64;
    float s = 0.f, q = 0.f;
    for (int b = b0; b < b0 + 64; ++b) {
        float nv = bnum[f];
        for (int k = 0; k < NUMF; ++k) nv += num[b * NUMF + k] * Wn[k * DIM + f];
        s += nv; q += nv * nv;
    }
    int idx = (blockIdx.x * 2 + g) * 128 + f;
    psum[idx] = s; psq[idx] = q;
}

__global__ void bn_stats2(const float* __restrict__ psum, const float* __restrict__ psq,
                          float* __restrict__ mu, float* __restrict__ rstd) {
    int f = threadIdx.x;
    float s = 0.f, q = 0.f;
    for (int i = 0; i < 128; ++i) { s += psum[i * 128 + f]; q += psq[i * 128 + f]; }
    float m = s * (1.0f / BATCH);
    float v = q * (1.0f / BATCH) - m * m;
    mu[f] = m;
    rstd[f] = rsqrtf(v + EPS);
}

// ---------------- fused transformer + head ----------------
__global__ __launch_bounds__(64, 1) void tx_kernel(
    const float* __restrict__ x,   const float* __restrict__ gcn,
    const float* __restrict__ num,
    const float* __restrict__ bq,  const float* __restrict__ bk,
    const float* __restrict__ bv,  const float* __restrict__ bo,
    const float* __restrict__ ln1g, const float* __restrict__ ln1b,
    const float* __restrict__ b1,  const float* __restrict__ b2,
    const float* __restrict__ ln2g, const float* __restrict__ ln2b,
    const float* __restrict__ Wn,  const float* __restrict__ bnum,
    const float* __restrict__ bng, const float* __restrict__ bnb,
    const float* __restrict__ Wf,  const float* __restrict__ bf,
    const float* __restrict__ pe,  const float* __restrict__ mu,
    const float* __restrict__ rstd, const _Float16* __restrict__ wh,
    float* __restrict__ out) {
    __shared__ float    curS[2][SEQ * DIM];
    __shared__ float    accS[2][SEQ * DIM];
    __shared__ _Float16 planeS[2][3 * SEQ * DIM];   // q/k/v f16, plane0 doubles as act f16
    __shared__ float    scoreS[2][256];
    __shared__ _Float16 attnS[2][256];
    __shared__ _Float16 hbufS[2][16 * 32];
    // async-staged FFN weights, shared by both waves: [buf][W1|W2][8 frags x 512 halfs]
    __shared__ _Float16 stageS[2][2][4096];

    const int wave = threadIdx.x >> 5;
    const int lane = threadIdx.x & 31;
    const int b    = blockIdx.x * 2 + wave;

    float*    cur   = curS[wave];
    float*    acc   = accS[wave];
    _Float16* qp    = planeS[wave];            // plane 0 (also act f16)
    _Float16* kp    = planeS[wave] + 2048;     // plane 1
    _Float16* vp    = planeS[wave] + 4096;     // plane 2
    float*    score = scoreS[wave];
    _Float16* attn  = attnS[wave];
    _Float16* hbuf  = hbufS[wave];

    const int n  = lane & 15;
    const int mb = (lane >> 4) * 8;

    // load x + PE (x is stream-once: non-temporal)
    for (int i = lane; i < SEQ * DIM; i += 32) {
        float v = __builtin_nontemporal_load(&x[(size_t)b * SEQ * DIM + i]) + pe[i];
        cur[i] = v;
        qp[i]  = (_Float16)v;
    }
    __syncthreads();

    for (int l = 0; l < 2; ++l) {
        v16h areg[4];
#pragma unroll
        for (int kt = 0; kt < 4; ++kt) areg[kt] = load_a_frag(qp, lane, kt * 32, DIM);

        // q/k/v (f16 outputs only); q overwrites act plane but A is in regs
        gemm128(areg, wh + HV + l * 16384, bv + l * DIM, nullptr, vp, lane);
        gemm128(areg, wh + HK + l * 16384, bk + l * DIM, nullptr, kp, lane);
        gemm128(areg, wh + HQ + l * 16384, bq + l * DIM, nullptr, qp, lane);
        __syncthreads();

        // attention: 8 "heads" from the raw reshape, each 16x16, K padded to 32
        for (int h = 0; h < 8; ++h) {
            int m    = lane & 15;
            int koff = (lane >> 4) * 8;
            int kb   = (lane >> 4) * 16;
            v16h a, bb;
#pragma unroll
            for (int e = 0; e < 16; ++e) {
                int k = koff + (e < 8 ? e : e + 8);
                a[e] = (k < 16) ? qp[(2 * h + (m >> 3)) * DIM + (m & 7) * 16 + k]
                                : (_Float16)0.0f;
            }
#pragma unroll
            for (int e = 0; e < 16; ++e) {
                int k = koff + (e < 8 ? e : e + 8);   // B[k][n] = k_h[n][k]  (K^T)
                bb[e] = (k < 16) ? kp[(2 * h + (n >> 3)) * DIM + (n & 7) * 16 + k]
                                 : (_Float16)0.0f;
            }
            v8f sc = {};
            sc = wmma16(a, bb, sc);
#pragma unroll
            for (int r = 0; r < 8; ++r) score[(mb + r) * 16 + n] = sc[r];
            __syncthreads();

            if (lane < 16) {
                float row[16], mx = -1e30f;
#pragma unroll
                for (int j = 0; j < 16; ++j) {
                    row[j] = score[lane * 16 + j] * 0.25f;  // DH^-0.5
                    mx = fmaxf(mx, row[j]);
                }
                float s = 0.f;
#pragma unroll
                for (int j = 0; j < 16; ++j) { row[j] = expf(row[j] - mx); s += row[j]; }
                float inv = 1.0f / s;
#pragma unroll
                for (int j = 0; j < 16; ++j)
                    attn[lane * 16 + j] = (_Float16)(row[j] * inv);
            }
            __syncthreads();

#pragma unroll
            for (int e = 0; e < 16; ++e) {
                int k = koff + (e < 8 ? e : e + 8);
                a[e] = (k < 16) ? attn[m * 16 + k] : (_Float16)0.0f;
            }
#pragma unroll
            for (int e = 0; e < 16; ++e) {
                int k = koff + (e < 8 ? e : e + 8);   // B[k][n] = v_h[k][n]
                bb[e] = (k < 16) ? vp[(2 * h + (k >> 3)) * DIM + (k & 7) * 16 + n]
                                 : (_Float16)0.0f;
            }
            v8f cx = {};
            cx = wmma16(a, bb, cx);
#pragma unroll
            for (int r = 0; r < 8; ++r) {
                int mm = mb + r;
                acc[(2 * h + (mm >> 3)) * DIM + (mm & 7) * 16 + n] = cx[r];
            }
            __syncthreads();
        }

        // ctx -> f16, then @Wo + bo -> acc (f32)
        for (int i = lane; i < SEQ * DIM; i += 32) qp[i] = (_Float16)acc[i];
        __syncthreads();
#pragma unroll
        for (int kt = 0; kt < 4; ++kt) areg[kt] = load_a_frag(qp, lane, kt * 32, DIM);
        gemm128(areg, wh + HO + l * 16384, bo + l * DIM, acc, nullptr, lane);
        __syncthreads();
        ln_res(acc, cur, ln1g + l * DIM, ln1b + l * DIM, qp, lane);
        __syncthreads();

        // ---------------- FFN with async-staged weights ----------------
        v16h ah[4];
#pragma unroll
        for (int kt = 0; kt < 4; ++kt) ah[kt] = load_a_frag(qp, lane, kt * 32, DIM);
        v8f ff[8];
#pragma unroll
        for (int i = 0; i < 8; ++i) ff[i] = (v8f){};

        const _Float16* w1 = wh + H1 + l * 131072;
        const _Float16* w2 = wh + H2 + l * 131072;
        const float*    b1l = b1 + l * HID;

        // stage chunk hc of W1 (8KB contiguous) / W2 (8 x 1KB strided) into stageS[buf]
        auto stage_chunk = [&](int hc, int buf) {
            if (wave == 0) {
                const char* src = (const char*)(w1 + (size_t)hc * 4096);
                unsigned dst = (unsigned)(uintptr_t)&stageS[buf][0][0];
#pragma unroll
                for (int i = 0; i < 16; ++i)
                    async_ld_b128(dst + i * 512 + lane * 16, src + i * 512 + lane * 16);
            } else {
                unsigned dst = (unsigned)(uintptr_t)&stageS[buf][1][0];
#pragma unroll
                for (int p = 0; p < 8; ++p) {
                    const char* src = (const char*)(w2 + ((size_t)(p * 32 + hc)) * 512);
#pragma unroll
                    for (int j = 0; j < 2; ++j)
                        async_ld_b128(dst + p * 1024 + j * 512 + lane * 16,
                                      src + j * 512 + lane * 16);
                }
            }
        };

        stage_chunk(0, 0);
        wait_async0();
        __syncthreads();

        for (int hc = 0; hc < 32; ++hc) {
            int cb = hc & 1;
            if (hc < 31) stage_chunk(hc + 1, cb ^ 1);   // overlap with compute

            const _Float16* s1 = &stageS[cb][0][0];
            const _Float16* s2 = &stageS[cb][1][0];
#pragma unroll
            for (int half = 0; half < 2; ++half) {
                int nt = hc * 2 + half;
                v8f c = {};
#pragma unroll
                for (int kt = 0; kt < 4; ++kt)
                    c = wmma16(ah[kt], load_b_frag_lds(s1, lane, half * 4 + kt), c);
                float bvv = b1l[nt * 16 + n];
#pragma unroll
                for (int r = 0; r < 8; ++r) {
                    float t = fmaxf(c[r] + bvv, 0.0f);
                    hbuf[(mb + r) * 32 + half * 16 + n] = (_Float16)t;
                }
            }
            __syncthreads();
            v16h a2 = load_a_frag(hbuf, lane, 0, 32);
#pragma unroll
            for (int nt2 = 0; nt2 < 8; ++nt2)
                ff[nt2] = wmma16(a2, load_b_frag_lds(s2, lane, nt2), ff[nt2]);

            wait_async0();        // next buffer staged
            __syncthreads();      // publish to both waves; also guards hbuf reuse
        }
#pragma unroll
        for (int nt2 = 0; nt2 < 8; ++nt2) {
            float bvv = b2[l * DIM + nt2 * 16 + n];
#pragma unroll
            for (int r = 0; r < 8; ++r)
                acc[(mb + r) * DIM + nt2 * 16 + n] = ff[nt2][r] + bvv;
        }
        __syncthreads();
        ln_res(acc, cur, ln2g + l * DIM, ln2b + l * DIM, qp, lane);
        __syncthreads();
    }

    // ---------------- head ----------------
    float dot = 0.f;
    for (int i = lane; i < SEQ * DIM; i += 32) dot += cur[i] * Wf[i];
    for (int i = lane; i < GCN; i += 32) dot += gcn[(size_t)b * GCN + i] * Wf[2048 + i];
    for (int f = lane; f < DIM; f += 32) {
        float nv = bnum[f];
        for (int k = 0; k < NUMF; ++k) nv += num[(size_t)b * NUMF + k] * Wn[k * DIM + f];
        nv = (nv - mu[f]) * rstd[f] * bng[f] + bnb[f];
        dot += nv * Wf[2048 + GCN + f];
    }
    dot = wave_sum(dot);
    if (lane == 0)
        __builtin_nontemporal_store(1.0f / (1.0f + expf(-(dot + bf[0]))), &out[b]);
}

// ---------------- launch ----------------
extern "C" void kernel_launch(void* const* d_in, const int* in_sizes, int n_in,
                              void* d_out, int out_size, void* d_ws, size_t ws_size,
                              hipStream_t stream) {
    (void)in_sizes; (void)n_in; (void)out_size; (void)ws_size;
    const float* x     = (const float*)d_in[0];
    const float* gcn   = (const float*)d_in[1];
    const float* num   = (const float*)d_in[2];
    const float* Wq    = (const float*)d_in[3];
    const float* bq    = (const float*)d_in[4];
    const float* Wk    = (const float*)d_in[5];
    const float* bk    = (const float*)d_in[6];
    const float* Wv    = (const float*)d_in[7];
    const float* bvp   = (const float*)d_in[8];
    const float* Wo    = (const float*)d_in[9];
    const float* bo    = (const float*)d_in[10];
    const float* ln1g  = (const float*)d_in[11];
    const float* ln1b  = (const float*)d_in[12];
    const float* W1    = (const float*)d_in[13];
    const float* b1    = (const float*)d_in[14];
    const float* W2    = (const float*)d_in[15];
    const float* b2    = (const float*)d_in[16];
    const float* ln2g  = (const float*)d_in[17];
    const float* ln2b  = (const float*)d_in[18];
    const float* Wn    = (const float*)d_in[19];
    const float* bnum  = (const float*)d_in[20];
    const float* bng   = (const float*)d_in[21];
    const float* bnb   = (const float*)d_in[22];
    const float* Wf    = (const float*)d_in[23];
    const float* bf    = (const float*)d_in[24];

    char* ws = (char*)d_ws;
    float*    pe   = (float*)(ws + OFF_PE);
    float*    mu   = (float*)(ws + OFF_MU);
    float*    rstd = (float*)(ws + OFF_RSTD);
    float*    psum = (float*)(ws + OFF_PSUM);
    float*    psq  = (float*)(ws + OFF_PSQ);
    _Float16* wh   = (_Float16*)(ws + OFF_W);

    pe_kernel<<<(SEQ * DIM + 255) / 256, 256, 0, stream>>>(pe);

    cvt_frag<<<(2 * DIM * DIM + 255) / 256, 256, 0, stream>>>(Wq, wh + HQ, DIM, DIM, 2 * DIM * DIM);
    cvt_frag<<<(2 * DIM * DIM + 255) / 256, 256, 0, stream>>>(Wk, wh + HK, DIM, DIM, 2 * DIM * DIM);
    cvt_frag<<<(2 * DIM * DIM + 255) / 256, 256, 0, stream>>>(Wv, wh + HV, DIM, DIM, 2 * DIM * DIM);
    cvt_frag<<<(2 * DIM * DIM + 255) / 256, 256, 0, stream>>>(Wo, wh + HO, DIM, DIM, 2 * DIM * DIM);
    cvt_frag<<<(2 * DIM * HID + 255) / 256, 256, 0, stream>>>(W1, wh + H1, DIM, HID, 2 * DIM * HID);
    cvt_frag<<<(2 * HID * DIM + 255) / 256, 256, 0, stream>>>(W2, wh + H2, HID, DIM, 2 * HID * DIM);

    bn_stats1<<<64, 256, 0, stream>>>(num, Wn, bnum, psum, psq);
    bn_stats2<<<1, 128, 0, stream>>>(psum, psq, mu, rstd);

    tx_kernel<<<BATCH / 2, 64, 0, stream>>>(
        x, gcn, num, bq, bk, bvp, bo, ln1g, ln1b, b1, b2, ln2g, ln2b,
        Wn, bnum, bng, bnb, Wf, bf, pe, mu, rstd, wh, (float*)d_out);
}
